// MSARowAttentionWithBias_18330920419586
// MI455X (gfx1250) — matrix operand order repaired and checked by reading, not compiled, not run on hardware
//
#include <hip/hip_runtime.h>
#include <hip/hip_bf16.h>

typedef __attribute__((ext_vector_type(16))) __bf16 v16bf;
typedef __attribute__((ext_vector_type(8)))  __bf16 v8bf;
typedef __attribute__((ext_vector_type(8)))  float  v8f;

#define DEV __device__ __forceinline__

DEV float wred_sum(float v) {
#pragma unroll
  for (int m = 16; m >= 1; m >>= 1) v += __shfl_xor(v, m, 32);
  return v;
}
DEV float wred_max(float v) {
#pragma unroll
  for (int m = 16; m >= 1; m >>= 1) v = fmaxf(v, __shfl_xor(v, m, 32));
  return v;
}

// ---------------------------------------------------------------------------
// WMMA fragment loaders, matching CDNA5 ISA 7.12.2 (wave32).
// A (16x32 bf16, M x K): lane&15 = row M; lanes 0-15 hold K {0..7,16..23},
// lanes 16-31 hold K {8..15,24..31}; each half is 8 contiguous bf16.
DEV v16bf ldA(const __bf16* base, int stride, int lane) {
  const __bf16* p = base + (size_t)(lane & 15) * stride + ((lane >> 4) << 3);
  union { v16bf v; v8bf h[2]; } u;
  u.h[0] = *(const v8bf*)p;
  u.h[1] = *(const v8bf*)(p + 16);
  return u.v;
}
// B (32x16 bf16, K x N) from N-major storage BT[col][k]: lane&15 = col N;
// lanes 0-15 hold K {0..7,16..23}, lanes 16-31 hold K {8..15,24..31}.
DEV v16bf ldB(const __bf16* base, int stride, int lane) {
  const __bf16* p = base + (size_t)(lane & 15) * stride + ((lane >> 4) << 3);
  union { v16bf v; v8bf h[2]; } u;
  u.h[0] = *(const v8bf*)p;
  u.h[1] = *(const v8bf*)(p + 16);
  return u.v;
}

DEV v8f wmma_bf16(v16bf a, v16bf b, v8f c) {
  return __builtin_amdgcn_wmma_f32_16x16x32_bf16(false, a, false, b, (short)0, c,
                                                 false, false);
}

// ---------------------------------------------------------------------------
// Kernel 1: LayerNorm over last dim (256) of msa -> bf16. One wave per row.
__global__ void k_ln_msa(const float* __restrict__ x, const float* __restrict__ g,
                         const float* __restrict__ b, __bf16* __restrict__ out,
                         int rows) {
  int lane = threadIdx.x & 31;
  int row  = blockIdx.x * 8 + (threadIdx.x >> 5);
  if (row >= rows) return;
  const float* p = x + (size_t)row * 256;
  float v[8];
  float s = 0.f;
#pragma unroll
  for (int e = 0; e < 8; ++e) { v[e] = p[lane + 32 * e]; s += v[e]; }
  float mean = wred_sum(s) * (1.f / 256.f);
  float vs = 0.f;
#pragma unroll
  for (int e = 0; e < 8; ++e) { float d = v[e] - mean; vs += d * d; }
  float inv = rsqrtf(wred_sum(vs) * (1.f / 256.f) + 1e-5f);
  __bf16* o = out + (size_t)row * 256;
#pragma unroll
  for (int e = 0; e < 8; ++e) {
    int k = lane + 32 * e;
    o[k] = (__bf16)((v[e] - mean) * inv * g[k] + b[k]);
  }
}

// ---------------------------------------------------------------------------
// Kernel 2: transpose + fp32->bf16 weight (K x C) -> WT (C x K)
__global__ void k_transw(const float* __restrict__ W, __bf16* __restrict__ WT,
                         int K, int C) {
  int idx = blockIdx.x * 256 + threadIdx.x;
  if (idx >= K * C) return;
  int c = idx / K, k = idx - c * K;
  WT[idx] = (__bf16)W[(size_t)k * C + c];
}

// ---------------------------------------------------------------------------
// Kernel 3: generic GEMM  C[M,N=256] = A[M,K](bf16) * BT[256,K](bf16), K%32==0.
// Block = 256 thr = 8 waves; wave computes 32(M) x 64(N); block covers 128x128.
// 8 accumulators (64 VGPRs) -- fits in registers, no spills; 0.75KB per WMMA.
// Epilogue modes:
//  0: bf16 <- (acc + bias?)*scale          (ksw, k)
//  1: f32  <- (acc + bias?)*scale          (qsw, final output)
//  2: bf16 <- acc * seqweight[(i*8+h)*256+n]    (q)
//  3: bf16 transposed store vT[(n*256+c)*384+j] (v)
//  4: bf16 <- sigmoid(acc + bias)               (gate)
__global__ __launch_bounds__(256)
void k_gemm(const __bf16* __restrict__ A, const __bf16* __restrict__ BT,
            int M, int K, int N, const float* __restrict__ bias,
            float scale, int mode, void* __restrict__ out,
            const float* __restrict__ sw) {
  int lane = threadIdx.x & 31;
  int wave = threadIdx.x >> 5;
  int m0 = blockIdx.x * 128 + (wave & 3) * 32;
  int n0 = blockIdx.y * 128 + (wave >> 2) * 64;
  if (m0 >= M) return;
  v8f acc[8] = {};
  for (int k0 = 0; k0 < K; k0 += 32) {
    v16bf a0 = ldA(A + (size_t)m0 * K + k0, K, lane);
    v16bf a1 = ldA(A + (size_t)(m0 + 16) * K + k0, K, lane);
#pragma unroll
    for (int t = 0; t < 4; ++t) {
      v16bf b = ldB(BT + (size_t)(n0 + t * 16) * K + k0, K, lane);
      acc[t]     = wmma_bf16(a0, b, acc[t]);
      acc[4 + t] = wmma_bf16(a1, b, acc[4 + t]);
    }
  }
#pragma unroll
  for (int a = 0; a < 2; ++a) {
#pragma unroll
    for (int t = 0; t < 4; ++t) {
#pragma unroll
      for (int e = 0; e < 8; ++e) {
        int r = m0 + a * 16 + e + ((lane >> 4) << 3);
        int c = n0 + t * 16 + (lane & 15);
        float x = acc[a * 4 + t][e];
        if (mode == 0) {
          if (bias) x += bias[c];
          ((__bf16*)out)[(size_t)r * N + c] = (__bf16)(x * scale);
        } else if (mode == 1) {
          if (bias) x += bias[c];
          ((float*)out)[(size_t)r * N + c] = x * scale;
        } else if (mode == 2) {
          int i = r % 384, n = r / 384, h = c >> 5;
          x *= sw[((size_t)i * 8 + h) * 256 + n];
          ((__bf16*)out)[(size_t)r * N + c] = (__bf16)x;
        } else if (mode == 3) {
          int n = r / 384, j = r % 384;
          ((__bf16*)out)[((size_t)n * 256 + c) * 384 + j] = (__bf16)x;
        } else {
          x += bias[c];
          ((__bf16*)out)[(size_t)r * N + c] = (__bf16)(1.f / (1.f + __expf(-x)));
        }
      }
    }
  }
}

// ---------------------------------------------------------------------------
// Kernel 4: seq-weight logits: swlog[(i*8+h)*256+n] = qsw[i,h,:] . ksw[n,i,h,:]
__global__ void k_swdot(const float* __restrict__ qsw, const __bf16* __restrict__ ksw,
                        float* __restrict__ swlog) {
  int idx = blockIdx.x * 256 + threadIdx.x;
  if (idx >= 256 * 384 * 8) return;
  int h = idx & 7;
  int i = (idx >> 3) % 384;
  int n = idx / (384 * 8);
  const float*  qp = qsw + (size_t)i * 256 + h * 32;
  const __bf16* kp = ksw + ((size_t)n * 384 + i) * 256 + h * 32;
  float s = 0.f;
#pragma unroll
  for (int d = 0; d < 32; ++d) s += qp[d] * (float)kp[d];
  swlog[((size_t)i * 8 + h) * 256 + n] = s;
}

// Kernel 5: softmax over rows of 256 (in place, f32). One wave per row.
__global__ void k_softmax256(float* __restrict__ buf, int rows) {
  int lane = threadIdx.x & 31;
  int row  = blockIdx.x * 8 + (threadIdx.x >> 5);
  if (row >= rows) return;
  float* p = buf + (size_t)row * 256;
  float v[8], mx = -3.4e38f;
#pragma unroll
  for (int e = 0; e < 8; ++e) { v[e] = p[lane + 32 * e]; mx = fmaxf(mx, v[e]); }
  mx = wred_max(mx);
  float s = 0.f;
#pragma unroll
  for (int e = 0; e < 8; ++e) { v[e] = __expf(v[e] - mx); s += v[e]; }
  float r = 1.f / wred_sum(s);
#pragma unroll
  for (int e = 0; e < 8; ++e) p[lane + 32 * e] = v[e] * r;
}

// ---------------------------------------------------------------------------
// Kernel 6: attention logits: logits[h][i][j] = sum_{n,d} qs[n,i,h,d]*ks[n,j,h,d]
// Wave = 16(i) x 64(j) tile; K loop over n with K=32 (=D) per WMMA step.
__global__ __launch_bounds__(256)
void k_gemm_attn(const __bf16* __restrict__ qs,
                 const __bf16* __restrict__ ks,
                 float* __restrict__ logits) {
  int lane = threadIdx.x & 31;
  int wave = threadIdx.x >> 5;
  int i0 = blockIdx.x * 128 + wave * 16;
  int j0 = blockIdx.y * 64;
  int h  = blockIdx.z;
  v8f acc[4] = {};
  for (int n = 0; n < 256; ++n) {
    const __bf16* qb = qs + ((size_t)n * 384) * 256 + (size_t)h * 32;
    const __bf16* kb = ks + ((size_t)n * 384) * 256 + (size_t)h * 32;
    v16bf a = ldA(qb + (size_t)i0 * 256, 256, lane);
#pragma unroll
    for (int t = 0; t < 4; ++t) {
      v16bf b = ldB(kb + (size_t)(j0 + t * 16) * 256, 256, lane);
      acc[t] = wmma_bf16(a, b, acc[t]);
    }
  }
  float* Lp = logits + (size_t)h * 384 * 384;
#pragma unroll
  for (int t = 0; t < 4; ++t)
#pragma unroll
    for (int e = 0; e < 8; ++e) {
      int i = i0 + e + ((lane >> 4) << 3);
      int j = j0 + t * 16 + (lane & 15);
      Lp[(size_t)i * 384 + j] = acc[t][e];
    }
}

// Kernel 7: pair bias: logits[h][i][j] += LN(pair[i,j,:128]) . wb[:,h]
// One wave per (i,j); 4 elements per lane.
__global__ void k_pairbias(const float* __restrict__ pair, const float* __restrict__ g,
                           const float* __restrict__ b, const float* __restrict__ wb,
                           float* __restrict__ logits) {
  int lane = threadIdx.x & 31;
  int wid  = blockIdx.x * 8 + (threadIdx.x >> 5);  // i*384 + j
  if (wid >= 384 * 384) return;
  const float* pp = pair + (size_t)wid * 128;
  float x[4], s = 0.f;
#pragma unroll
  for (int e = 0; e < 4; ++e) { x[e] = pp[lane * 4 + e]; s += x[e]; }
  float mean = wred_sum(s) * (1.f / 128.f);
  float vs = 0.f;
#pragma unroll
  for (int e = 0; e < 4; ++e) { float d = x[e] - mean; vs += d * d; }
  float inv = rsqrtf(wred_sum(vs) * (1.f / 128.f) + 1e-5f);
  float ph[8] = {};
#pragma unroll
  for (int e = 0; e < 4; ++e) {
    int k = lane * 4 + e;
    float xn = (x[e] - mean) * inv * g[k] + b[k];
#pragma unroll
    for (int h = 0; h < 8; ++h) ph[h] += xn * wb[(size_t)k * 8 + h];
  }
#pragma unroll
  for (int h = 0; h < 8; ++h) ph[h] = wred_sum(ph[h]);
  if (lane == 0)
#pragma unroll
    for (int h = 0; h < 8; ++h) logits[(size_t)h * 384 * 384 + wid] += ph[h];
}

// Kernel 8: softmax over j (rows of 384), f32 -> bf16. One wave per (h,i).
__global__ void k_softmax384(const float* __restrict__ logits,
                             __bf16* __restrict__ attn, int rows) {
  int lane = threadIdx.x & 31;
  int row  = blockIdx.x * 8 + (threadIdx.x >> 5);
  if (row >= rows) return;
  const float* p = logits + (size_t)row * 384;
  float v[12], mx = -3.4e38f;
#pragma unroll
  for (int e = 0; e < 12; ++e) { v[e] = p[lane + 32 * e]; mx = fmaxf(mx, v[e]); }
  mx = wred_max(mx);
  float s = 0.f;
#pragma unroll
  for (int e = 0; e < 12; ++e) { v[e] = __expf(v[e] - mx); s += v[e]; }
  float r = 1.f / wred_sum(s);
  __bf16* o = attn + (size_t)row * 384;
#pragma unroll
  for (int e = 0; e < 12; ++e) o[lane + 32 * e] = (__bf16)(v[e] * r);
}

// ---------------------------------------------------------------------------
// Kernel 9: og[n,i,h*32+d] = gate * sum_j attn[h][i][j] * vT[(n*8+h)*32+d][j]
// Wave = 32(i) x 32(d) tile for one (n,h); K=j in steps of 32.
__global__ __launch_bounds__(256)
void k_gemm_av(const __bf16* __restrict__ attn, const __bf16* __restrict__ vT,
               const __bf16* __restrict__ gate, __bf16* __restrict__ og) {
  int lane = threadIdx.x & 31;
  int task = blockIdx.x * 8 + (threadIdx.x >> 5);  // 12*8*256 tasks
  if (task >= 12 * 8 * 256) return;
  int it = task % 12;
  int h  = (task / 12) & 7;
  int n  = task / 96;
  int i0 = it * 32;
  const __bf16* Ab = attn + (size_t)h * 384 * 384 + (size_t)i0 * 384;
  const __bf16* Bb = vT + ((size_t)(n * 8 + h) * 32) * 384;
  v8f acc[4] = {};
  for (int k0 = 0; k0 < 384; k0 += 32) {
    v16bf a0 = ldA(Ab + k0, 384, lane);
    v16bf a1 = ldA(Ab + (size_t)16 * 384 + k0, 384, lane);
#pragma unroll
    for (int t = 0; t < 2; ++t) {
      v16bf b = ldB(Bb + (size_t)(t * 16) * 384 + k0, 384, lane);
      acc[t]     = wmma_bf16(a0, b, acc[t]);
      acc[2 + t] = wmma_bf16(a1, b, acc[2 + t]);
    }
  }
#pragma unroll
  for (int a = 0; a < 2; ++a)
#pragma unroll
    for (int t = 0; t < 2; ++t)
#pragma unroll
      for (int e = 0; e < 8; ++e) {
        int i = i0 + a * 16 + e + ((lane >> 4) << 3);
        int c = h * 32 + t * 16 + (lane & 15);
        size_t idx = ((size_t)n * 384 + i) * 256 + c;
        og[idx] = (__bf16)(acc[a * 2 + t][e] * (float)gate[idx]);
      }
}

// ---------------------------------------------------------------------------
extern "C" void kernel_launch(void* const* d_in, const int* in_sizes, int n_in,
                              void* d_out, int out_size, void* d_ws, size_t ws_size,
                              hipStream_t stream) {
  const float* msa      = (const float*)d_in[0];
  const float* pair     = (const float*)d_in[1];
  const float* ln_msa_g = (const float*)d_in[2];
  const float* ln_msa_b = (const float*)d_in[3];
  const float* ln_pr_g  = (const float*)d_in[4];
  const float* ln_pr_b  = (const float*)d_in[5];
  const float* sw_wq    = (const float*)d_in[6];
  const float* sw_bq    = (const float*)d_in[7];
  const float* sw_wk    = (const float*)d_in[8];
  const float* sw_bk    = (const float*)d_in[9];
  const float* wq       = (const float*)d_in[10];
  const float* wk       = (const float*)d_in[11];
  const float* wv       = (const float*)d_in[12];
  const float* wb       = (const float*)d_in[13];
  const float* wg       = (const float*)d_in[14];
  const float* bg       = (const float*)d_in[15];
  const float* wo       = (const float*)d_in[16];
  const float* bo       = (const float*)d_in[17];
  float* out = (float*)d_out;

  const int L = 384, H = 8, DM = 256;
  const int ML = 256 * L;                 // 98304 rows of m
  const float inv_sqrt_d = 0.17677669529663687f;  // 1/sqrt(32)

  char* w = (char*)d_ws;
  auto alloc = [&](size_t bytes) -> char* {
    char* p = w;
    w += (bytes + 255) & ~(size_t)255;
    return p;
  };
  __bf16* m_bf    = (__bf16*)alloc((size_t)ML * DM * 2);
  __bf16* wT_swq  = (__bf16*)alloc(256 * 256 * 2);
  __bf16* wT_swk  = (__bf16*)alloc(256 * 256 * 2);
  __bf16* wT_q    = (__bf16*)alloc(256 * 256 * 2);
  __bf16* wT_k    = (__bf16*)alloc(256 * 256 * 2);
  __bf16* wT_v    = (__bf16*)alloc(256 * 256 * 2);
  __bf16* wT_g    = (__bf16*)alloc(256 * 256 * 2);
  __bf16* wT_o    = (__bf16*)alloc(256 * 256 * 2);
  __bf16* ksw_bf  = (__bf16*)alloc((size_t)ML * DM * 2);
  float*  qsw_f   = (float*) alloc((size_t)L * DM * 4);
  float*  swbuf   = (float*) alloc((size_t)L * H * 256 * 4);
  __bf16* qs_bf   = (__bf16*)alloc((size_t)ML * DM * 2);
  __bf16* ks_bf   = (__bf16*)alloc((size_t)ML * DM * 2);
  __bf16* vT_bf   = (__bf16*)alloc((size_t)ML * DM * 2);
  __bf16* gate_bf = (__bf16*)alloc((size_t)ML * DM * 2);
  float*  logits  = (float*) alloc((size_t)H * L * L * 4);
  __bf16* attn_bf = (__bf16*)alloc((size_t)H * L * L * 2);
  __bf16* og_bf   = (__bf16*)alloc((size_t)ML * DM * 2);

  // weights -> bf16 transposed (N x K)
  k_transw<<<256, 256, 0, stream>>>(sw_wq, wT_swq, 256, 256);
  k_transw<<<256, 256, 0, stream>>>(sw_wk, wT_swk, 256, 256);
  k_transw<<<256, 256, 0, stream>>>(wq, wT_q, 256, 256);
  k_transw<<<256, 256, 0, stream>>>(wk, wT_k, 256, 256);
  k_transw<<<256, 256, 0, stream>>>(wv, wT_v, 256, 256);
  k_transw<<<256, 256, 0, stream>>>(wg, wT_g, 256, 256);
  k_transw<<<256, 256, 0, stream>>>(wo, wT_o, 256, 256);

  // m = LN(msa), bf16
  k_ln_msa<<<ML / 8, 256, 0, stream>>>(msa, ln_msa_g, ln_msa_b, m_bf, ML);

  // ksw = m @ sw_wk + sw_bk   (bf16)
  k_gemm<<<dim3(ML / 128, 2), 256, 0, stream>>>(m_bf, wT_swk, ML, 256, 256,
                                                sw_bk, 1.f, 0, ksw_bf, nullptr);
  // qsw = (m[0:L] @ sw_wq + sw_bq) / sqrt(D)   (f32)
  k_gemm<<<dim3(L / 128, 2), 256, 0, stream>>>(m_bf, wT_swq, L, 256, 256,
                                               sw_bq, inv_sqrt_d, 1, qsw_f, nullptr);
  // seq-weight logits + softmax over n
  k_swdot<<<(256 * L * H) / 256, 256, 0, stream>>>(qsw_f, ksw_bf, swbuf);
  k_softmax256<<<(L * H) / 8, 256, 0, stream>>>(swbuf, L * H);

  // q = (m @ wq) * seq_weight ; k = (m @ wk)/sqrt(D) ; vT ; gate
  k_gemm<<<dim3(ML / 128, 2), 256, 0, stream>>>(m_bf, wT_q, ML, 256, 256,
                                                nullptr, 1.f, 2, qs_bf, swbuf);
  k_gemm<<<dim3(ML / 128, 2), 256, 0, stream>>>(m_bf, wT_k, ML, 256, 256,
                                                nullptr, inv_sqrt_d, 0, ks_bf, nullptr);
  k_gemm<<<dim3(ML / 128, 2), 256, 0, stream>>>(m_bf, wT_v, ML, 256, 256,
                                                nullptr, 1.f, 3, vT_bf, nullptr);
  k_gemm<<<dim3(ML / 128, 2), 256, 0, stream>>>(m_bf, wT_g, ML, 256, 256,
                                                bg, 1.f, 4, gate_bf, nullptr);

  // attn logits over (n,d), + pair bias, softmax over j
  k_gemm_attn<<<dim3(L / 128, L / 64, H), 256, 0, stream>>>(qs_bf, ks_bf, logits);
  k_pairbias<<<(L * L) / 8, 256, 0, stream>>>(pair, ln_pr_g, ln_pr_b, wb, logits);
  k_softmax384<<<(H * L) / 8, 256, 0, stream>>>(logits, attn_bf, H * L);

  // out = gate * (attn @ v)
  k_gemm_av<<<(12 * H * 256) / 8, 256, 0, stream>>>(attn_bf, vT_bf, gate_bf, og_bf);

  // final: out @ wo + bo  (f32 to d_out)
  k_gemm<<<dim3(ML / 128, 2), 256, 0, stream>>>(og_bf, wT_o, ML, 256, 256,
                                                bo, 1.f, 1, out, nullptr);
}
